// LSTMTemporalModel_249108103630
// MI455X (gfx1250) — matrix operand order repaired and checked by reading, not compiled, not run on hardware
//
#include <hip/hip_runtime.h>

// ---------------------------------------------------------------------------
// Fused 2-layer LSTM + dense head for MI455X (gfx1250), wave32 + bf16 WMMA.
//
// One persistent workgroup (16 waves) per 16-row batch tile. Per time step:
//   gates1(16x512) = [x_t | h1] (16x256, bf16) @ W1^T   (v_wmma_f32_16x16x32_bf16)
//   gates2(16x256) = [h1_t | h2] (16x192, bf16) @ W2^T
// Weight fragments live in VGPRs across all 128 steps; cell state fp32 regs.
// x_{t+1} is global-loaded at the top of each step via SGPR-base + 32-bit
// lane offset (GVS addressing: uniform part stays scalar, 1 VGPR of divergent
// state) and committed to LDS in the last phase, hiding HBM latency under 3
// phases of WMMA work. 4 barriers/step. A-fragments staged in pairs so peak
// pressure stays under the 256-VGPR ceiling (no spills).
// ---------------------------------------------------------------------------

typedef __bf16 bf16_t;
typedef __attribute__((ext_vector_type(16))) __bf16 v16bf;
typedef __attribute__((ext_vector_type(8)))  __bf16 v8bf;
typedef __attribute__((ext_vector_type(4)))  __bf16 v4bf;
typedef __attribute__((ext_vector_type(8)))  float  v8f;
typedef __attribute__((ext_vector_type(4)))  float  v4f;

#define B_TOT    2048
#define T_SEQ    128
#define D_IN     128
#define H1       128
#define G1       512   // 4*H1
#define H2       64
#define G2       256   // 4*H2
#define BT       16    // batch rows per workgroup
#define NTHREADS 512   // 16 waves

#define A1W 264        // 256 K + 8 pad (row = 528B, 16B aligned, conflict-free)
#define A2W 200        // 192 K + 8 pad (row = 400B, 16B aligned)

__device__ __forceinline__ float sigmoid_fast(float v) {
    return 1.0f / (1.0f + __expf(-v));
}
// tanh(x) = 1 - 2/(exp(2x)+1): branch-free, saturates correctly at +-inf.
__device__ __forceinline__ float tanh_fast(float v) {
    return 1.0f - 2.0f / (__expf(2.0f * v) + 1.0f);
}

// A fragment for v_wmma_f32_16x16x32_bf16 (contiguous k order; B uses the same):
// lane (m, half): elements 0..15 = K[kc*32 + half*16 + 0..15]
__device__ __forceinline__ v16bf load_a_frag(const bf16_t* row, int kc, int half) {
    const int base = kc * 32 + half * 16;
    v8bf lo = *(const v8bf*)(row + base);
    v8bf hi = *(const v8bf*)(row + base + 8);
    return __builtin_shufflevector(lo, hi, 0, 1, 2, 3, 4, 5, 6, 7,
                                           8, 9, 10, 11, 12, 13, 14, 15);
}

// B fragment (contiguous k order, as in A): lane (n = l&15, half) holds
// B[k = kc*32 + half*16 + j][n], j = 0..15. Combined K = [x-part | h-part];
// runs of 16 start at multiples of 16, never straddling the dx boundary.
__device__ __forceinline__ v16bf make_b_frag(const float* __restrict__ wx,
                                             const float* __restrict__ wh,
                                             int n, int dx, int dh, int ks) {
    const float* src = (ks < dx) ? (wx + (size_t)n * dx + ks)
                                 : (wh + (size_t)n * dh + (ks - dx));
    v16bf f;
#pragma unroll
    for (int j = 0; j < 16; ++j) f[j] = (bf16_t)src[j];
    return f;
}

__global__ __launch_bounds__(NTHREADS)
void lstm2_fused_kernel(const float* __restrict__ x,
                        const float* __restrict__ w_ih1, const float* __restrict__ w_hh1,
                        const float* __restrict__ b_ih1, const float* __restrict__ b_hh1,
                        const float* __restrict__ w_ih2, const float* __restrict__ w_hh2,
                        const float* __restrict__ b_ih2, const float* __restrict__ b_hh2,
                        const float* __restrict__ w_dense, const float* __restrict__ b_dense,
                        float* __restrict__ out)
{
    __shared__ bf16_t A1[BT][A1W];        // [x_t | h1] bf16, 8448 B
    __shared__ bf16_t A2[BT][A2W];        // [h1_t | h2] bf16, 6400 B
    __shared__ float  gates[BT][G1];      // reused by L1 (512 cols) and L2 (256 cols), 32 KB
    __shared__ float  h2last[BT][H2 + 1]; // final hidden state for dense head

    const int tid  = threadIdx.x;
    const int wave = tid >> 5;
    const int lane = tid & 31;
    const int m    = lane & 15;
    const int half = lane >> 4;
    const int b0   = blockIdx.x * BT;

    // ---- Load weight (B-matrix) fragments into registers, f32 -> bf16 ------
    // L1: wave owns gate cols [wave*32, wave*32+32), combined K = 256.
    v16bf b1[2][8];
#pragma unroll
    for (int nt = 0; nt < 2; ++nt) {
        const int n = wave * 32 + nt * 16 + m;
#pragma unroll
        for (int kc = 0; kc < 8; ++kc)
            b1[nt][kc] = make_b_frag(w_ih1, w_hh1, n, D_IN, H1, kc * 32 + half * 16);
    }
    // L2: wave owns gate cols [wave*16, wave*16+16), combined K = 192.
    v16bf b2[6];
    {
        const int n = wave * 16 + m;
#pragma unroll
        for (int kc = 0; kc < 6; ++kc)
            b2[kc] = make_b_frag(w_ih2, w_hh2, n, H1, H2, kc * 32 + half * 16);
    }
    // Per-column fused biases (b_ih + b_hh), broadcast across the M rows.
    const float bc1_0 = b_ih1[wave * 32 + m]      + b_hh1[wave * 32 + m];
    const float bc1_1 = b_ih1[wave * 32 + 16 + m] + b_hh1[wave * 32 + 16 + m];
    const float bc2   = b_ih2[wave * 16 + m]      + b_hh2[wave * 16 + m];

    // Cell states in registers (fixed thread<->element mapping per step).
    float c1r[4] = {0.f, 0.f, 0.f, 0.f};
    float c2r[2] = {0.f, 0.f};

    // Zero h-parts of the A buffers (h1_0 = h2_0 = 0).
    for (int i = tid; i < BT * A1W; i += NTHREADS) (&A1[0][0])[i] = (bf16_t)0.0f;
    for (int i = tid; i < BT * A2W; i += NTHREADS) (&A2[0][0])[i] = (bf16_t)0.0f;

    // ---- x staging: thread (wave=row b, lane) owns a contiguous float4. ----
    // Address = uniform SGPR part (x + b0*T*D + t*D, advanced scalar-side)
    //         + divergent loop-invariant 32-bit element offset (1 VGPR).
    const unsigned xvoff = (unsigned)wave * (T_SEQ * D_IN) + 4u * lane;
    const float* __restrict__ xblk = x + (size_t)b0 * (T_SEQ * D_IN);
    {   // preload x_0: coalesced b128 load, packed b64 LDS store
        v4f x0 = *(const v4f*)(xblk + xvoff);
        v4bf hv;
#pragma unroll
        for (int j = 0; j < 4; ++j) hv[j] = (bf16_t)x0[j];
        *(v4bf*)&A1[wave][4 * lane] = hv;
    }
    __syncthreads();

    for (int t = 0; t < T_SEQ; ++t) {
        // Issue x_{t+1} load unconditionally (index clamped on the last step;
        // the commit below is guarded instead). Consumed in the last phase,
        // ~3 barrier-separated phases later, so HBM latency is hidden.
        const int tnext = (t + 1 < T_SEQ) ? (t + 1) : t;
        v4f xnext = *(const v4f*)(xblk + (size_t)tnext * D_IN + xvoff);

        // -- Phase A: L1 gates = [x_t|h1] @ W1^T + bias (bf16 WMMA, K=256) ---
        {
            v8f acc0, acc1;
#pragma unroll
            for (int r = 0; r < 8; ++r) { acc0[r] = bc1_0; acc1[r] = bc1_1; }
#pragma unroll
            for (int g = 0; g < 4; ++g) {      // 4 groups of 2 K-chunks
                v16bf afr[2];
#pragma unroll
                for (int q = 0; q < 2; ++q)    // 4 back-to-back ds_load_b128
                    afr[q] = load_a_frag(&A1[m][0], g * 2 + q, half);
#pragma unroll
                for (int q = 0; q < 2; ++q) {  // 4 back-to-back v_wmma
                    acc0 = __builtin_amdgcn_wmma_f32_16x16x32_bf16(
                               false, afr[q], false, b1[0][g * 2 + q], (short)0, acc0, false, false);
                    acc1 = __builtin_amdgcn_wmma_f32_16x16x32_bf16(
                               false, afr[q], false, b1[1][g * 2 + q], (short)0, acc1, false, false);
                }
            }
            // C layout: VGPR r -> row (r + 8*half), col = m within the N-tile.
#pragma unroll
            for (int r = 0; r < 8; ++r) {
                gates[r + 8 * half][wave * 32 + m]      = acc0[r];
                gates[r + 8 * half][wave * 32 + 16 + m] = acc1[r];
            }
        }
        __syncthreads();

        // -- Phase B: L1 activations; h1 -> A1 (recurrent) and A2 (L2 input) -
        {
            const int j    = tid & 127;
            const int brow = tid >> 7;
#pragma unroll
            for (int it = 0; it < 4; ++it) {
                const int b = brow + it * 4;
                float ig = sigmoid_fast(gates[b][j]);
                float fg = sigmoid_fast(gates[b][128 + j]);
                float gg = tanh_fast(gates[b][256 + j]);
                float og = sigmoid_fast(gates[b][384 + j]);
                float c  = fg * c1r[it] + ig * gg;
                c1r[it]  = c;
                float h  = og * tanh_fast(c);
                bf16_t hb = (bf16_t)h;
                A1[b][D_IN + j] = hb;   // next step's recurrent input
                A2[b][j]        = hb;   // this step's L2 input
            }
        }
        __syncthreads();

        // -- Phase C: L2 gates = [h1_t|h2] @ W2^T + bias (K=192) -------------
        {
            v8f acc;
#pragma unroll
            for (int r = 0; r < 8; ++r) acc[r] = bc2;
#pragma unroll
            for (int g = 0; g < 3; ++g) {      // 3 groups of 2 K-chunks
                v16bf afr[2];
#pragma unroll
                for (int q = 0; q < 2; ++q)    // 4 back-to-back ds_load_b128
                    afr[q] = load_a_frag(&A2[m][0], g * 2 + q, half);
#pragma unroll
                for (int q = 0; q < 2; ++q)    // 2 back-to-back v_wmma
                    acc = __builtin_amdgcn_wmma_f32_16x16x32_bf16(
                              false, afr[q], false, b2[g * 2 + q], (short)0, acc, false, false);
            }
#pragma unroll
            for (int r = 0; r < 8; ++r)
                gates[r + 8 * half][wave * 16 + m] = acc[r];
        }
        __syncthreads();

        // -- Phase D: L2 activations; h2 -> A2; commit staged x_{t+1} --------
        {
            const int j    = tid & 63;
            const int brow = tid >> 6;   // 0..7
#pragma unroll
            for (int it = 0; it < 2; ++it) {
                const int b = brow + it * 8;
                float ig = sigmoid_fast(gates[b][j]);
                float fg = sigmoid_fast(gates[b][64 + j]);
                float gg = tanh_fast(gates[b][128 + j]);
                float og = sigmoid_fast(gates[b][192 + j]);
                float c  = fg * c2r[it] + ig * gg;
                c2r[it]  = c;
                float h  = og * tanh_fast(c);
                A2[b][H1 + j] = (bf16_t)h;
                if (t == T_SEQ - 1) h2last[b][j] = h;
            }
            if (t + 1 < T_SEQ) {
                v4bf hv;
#pragma unroll
                for (int j4 = 0; j4 < 4; ++j4) hv[j4] = (bf16_t)xnext[j4];
                *(v4bf*)&A1[wave][4 * lane] = hv;
            }
        }
        __syncthreads();
    }

    // ---- Dense head: out = relu(h2_last @ w_dense^T + b_dense) -------------
    {
        const int o    = tid & 63;
        const int brow = tid >> 6;
#pragma unroll
        for (int it = 0; it < 2; ++it) {
            const int b = brow + it * 8;
            float acc = b_dense[o];
#pragma unroll 8
            for (int k = 0; k < H2; ++k)
                acc += h2last[b][k] * w_dense[o * H2 + k];
            out[(size_t)(b0 + b) * H2 + o] = fmaxf(acc, 0.0f);
        }
    }
}

extern "C" void kernel_launch(void* const* d_in, const int* in_sizes, int n_in,
                              void* d_out, int out_size, void* d_ws, size_t ws_size,
                              hipStream_t stream) {
    const float* x       = (const float*)d_in[0];
    const float* w_ih1   = (const float*)d_in[1];
    const float* w_hh1   = (const float*)d_in[2];
    const float* b_ih1   = (const float*)d_in[3];
    const float* b_hh1   = (const float*)d_in[4];
    const float* w_ih2   = (const float*)d_in[5];
    const float* w_hh2   = (const float*)d_in[6];
    const float* b_ih2   = (const float*)d_in[7];
    const float* b_hh2   = (const float*)d_in[8];
    const float* w_dense = (const float*)d_in[9];
    const float* b_dense = (const float*)d_in[10];
    float* out = (float*)d_out;

    dim3 grid(B_TOT / BT);    // 128 workgroups, one 16-row batch tile each
    dim3 block(NTHREADS);     // 16 waves (wave32)
    hipLaunchKernelGGL(lstm2_fused_kernel, grid, block, 0, stream,
                       x, w_ih1, w_hh1, b_ih1, b_hh1,
                       w_ih2, w_hh2, b_ih2, b_hh2,
                       w_dense, b_dense, out);
}
